// CoherenceNetWithGCN_45466523795918
// MI455X (gfx1250) — compile-verified, matrix operand order checked by reading, not run on hardware
//
#include <hip/hip_runtime.h>

// ---------------------------------------------------------------------------
// B=4, S=4096, E=2048, C=4096, H=256, A=256.  bf16 WMMA 16x16x32, f32 accum.
//   K1: fp32 -> bf16 convert
//   K2: proj = attendee @ W^T + b          (pipelined WMMA NT-GEMM)
//   K3: flash-attention over keys          (TDM async V staging + WMMA)
//   K4: tanh([attender|ctx_s|ctx_e] @ W_lin^T + b_lin)  (pipelined WMMA)
// ---------------------------------------------------------------------------

typedef __attribute__((ext_vector_type(16))) __bf16 v16bf;
typedef __attribute__((ext_vector_type(8)))  float  v8f;

#if defined(__HIP_DEVICE_COMPILE__) && \
    __has_builtin(__builtin_amdgcn_tensor_load_to_lds)
#define HAVE_TDM 1
#else
#define HAVE_TDM 0
#endif

__device__ __forceinline__ __bf16 f2bf(float f) {
  union { float f; unsigned u; } a; a.f = f;
  unsigned r = a.u + 0x7FFFu + ((a.u >> 16) & 1u);   // round-to-nearest-even
  unsigned short s = (unsigned short)(r >> 16);
  __bf16 b;
  __builtin_memcpy(&b, &s, 2);
  return b;
}

__device__ __forceinline__ v8f zero8() {
  v8f z = {0.f, 0.f, 0.f, 0.f, 0.f, 0.f, 0.f, 0.f};
  return z;
}

// NT fragment load: lane L -> row (L&15), K-offsets (L>>4)*8 and +16.
__device__ __forceinline__ v16bf load_frag(const __bf16* base, int ld, int lane) {
  const int r  = lane & 15;
  const int kb = (lane >> 4) << 3;
  const __bf16* p = base + (size_t)r * ld + kb;
  union { uint4 q[2]; v16bf v; } u;
  u.q[0] = *reinterpret_cast<const uint4*>(p);
  u.q[1] = *reinterpret_cast<const uint4*>(p + 16);
  return u.v;
}

__device__ __forceinline__ v8f wmma_bf16(v16bf a, v16bf b, v8f c) {
  return __builtin_amdgcn_wmma_f32_16x16x32_bf16(false, a, false, b, (short)0, c,
                                                 false, false);
}

#if HAVE_TDM
typedef unsigned int u32x4 __attribute__((ext_vector_type(4)));
typedef int          i32x4 __attribute__((ext_vector_type(4)));
typedef int          i32x8 __attribute__((ext_vector_type(8)));

__device__ __forceinline__ unsigned lds_byte_off(const void* p) {
  return (unsigned)(unsigned long long)(
      (__attribute__((address_space(3))) const void*)p);
}

// 2-D TDM: load `rows` x `row_elems` bf16 tile (row stride `stride_elems`)
// from global to LDS.  D# per CDNA5 ISA.
__device__ __forceinline__ void tdm_load_2d(unsigned lds_off, const void* gsrc,
                                            unsigned rows, unsigned row_elems,
                                            unsigned stride_elems) {
  unsigned long long ga = (unsigned long long)gsrc;
  u32x4 g0;
  g0[0] = 1u;                                     // count=1, user descriptor
  g0[1] = lds_off;                                // lds_addr (bytes)
  g0[2] = (unsigned)(ga & 0xFFFFFFFFu);           // global_addr[31:0]
  g0[3] = (unsigned)((ga >> 32) & 0x01FFFFFFu) | (2u << 30);  // addr[56:32]|type=2
  i32x8 g1;
  g1[0] = (int)(1u << 16);                        // data_size=1 (2 bytes)
  g1[1] = (int)((row_elems & 0xFFFFu) << 16);     // tensor_dim0[15:0]
  g1[2] = (int)(((row_elems >> 16) & 0xFFFFu) | ((rows & 0xFFFFu) << 16));
  g1[3] = (int)(((rows >> 16) & 0xFFFFu) | ((row_elems & 0xFFFFu) << 16)); // tile_dim0
  g1[4] = (int)(rows & 0xFFFFu);                  // tile_dim1 (tile_dim2=0)
  g1[5] = (int)stride_elems;                      // tensor_dim0_stride[31:0]
  g1[6] = 0;
  g1[7] = 0;
  i32x4 z4 = {0, 0, 0, 0};
#if __clang_major__ >= 23
  i32x8 z8 = {0, 0, 0, 0, 0, 0, 0, 0};
  __builtin_amdgcn_tensor_load_to_lds(g0, g1, z4, z4, z8, 0);
#else
  __builtin_amdgcn_tensor_load_to_lds(g0, g1, z4, z4, 0);
#endif
}

__device__ __forceinline__ void wait_tensorcnt0() {
#if __has_builtin(__builtin_amdgcn_s_wait_tensorcnt)
  __builtin_amdgcn_s_wait_tensorcnt(0);
#else
  asm volatile("s_wait_tensorcnt 0x0" ::: "memory");
#endif
}
#endif  // HAVE_TDM

// ---------------------------------------------------------------------------
// K1: fp32 -> bf16
// ---------------------------------------------------------------------------
__global__ void cvt_bf16_kernel(const float* __restrict__ in,
                                __bf16* __restrict__ out, int n) {
  for (int i = blockIdx.x * blockDim.x + threadIdx.x; i < n;
       i += gridDim.x * blockDim.x)
    out[i] = f2bf(in[i]);
}

// ---------------------------------------------------------------------------
// K2: proj[row, h] = sum_k A[row,k]*W[h,k] + bias[h].  Software-pipelined.
// ---------------------------------------------------------------------------
__global__ __launch_bounds__(128) void gemm_proj_kernel(
    const __bf16* __restrict__ Amat, const __bf16* __restrict__ W,
    const float* __restrict__ bias, __bf16* __restrict__ out, int rows) {
  const int lane = threadIdx.x & 31;
  const int wv   = threadIdx.x >> 5;
  const int row0 = blockIdx.x * 64 + wv * 16;
  const int col0 = blockIdx.y * 64;
  if (row0 >= rows) return;

  v8f acc[4];
#pragma unroll
  for (int j = 0; j < 4; ++j) acc[j] = zero8();

  const __bf16* Ab = Amat + (size_t)row0 * 256;
  v16bf a_cur = load_frag(Ab, 256, lane);
  v16bf b_cur[4];
#pragma unroll
  for (int j = 0; j < 4; ++j)
    b_cur[j] = load_frag(W + (size_t)(col0 + j * 16) * 256, 256, lane);

#pragma unroll
  for (int kh = 0; kh < 8; ++kh) {
    v16bf a_nxt = a_cur;
    v16bf b_nxt[4];
    if (kh < 7) {  // prefetch next k-step before consuming current fragments
      a_nxt = load_frag(Ab + (kh + 1) * 32, 256, lane);
#pragma unroll
      for (int j = 0; j < 4; ++j)
        b_nxt[j] = load_frag(W + (size_t)(col0 + j * 16) * 256 + (kh + 1) * 32,
                             256, lane);
    }
#pragma unroll
    for (int j = 0; j < 4; ++j) acc[j] = wmma_bf16(a_cur, b_cur[j], acc[j]);
    a_cur = a_nxt;
#pragma unroll
    for (int j = 0; j < 4; ++j) b_cur[j] = (kh < 7) ? b_nxt[j] : b_cur[j];
  }

  const int hlf = lane >> 4, ln = lane & 15;
#pragma unroll
  for (int j = 0; j < 4; ++j) {
    const int col = col0 + j * 16 + ln;
    const float bb = bias[col];
#pragma unroll
    for (int r = 0; r < 8; ++r) {
      const int row = row0 + r + 8 * hlf;
      out[(size_t)row * 256 + col] = f2bf(acc[j][r] + bb);
    }
  }
}

// ---------------------------------------------------------------------------
// K3: flash attention, softmax over keys (axis n).  64 queries / block.
//   Q:[B,4096,256]  K:[B,N,256](proj)  V:[B,N,256]  mask:[B,N,4096] bool
//   TDM double-buffers raw V key-chunks into LDS; transpose pass builds V^T.
// ---------------------------------------------------------------------------
__global__ __launch_bounds__(128) void flash_attend_kernel(
    const __bf16* __restrict__ Q, const __bf16* __restrict__ K,
    const __bf16* __restrict__ V, const unsigned char* __restrict__ mask,
    __bf16* __restrict__ ctx, int N) {
  const int b    = blockIdx.y;
  const int c0   = blockIdx.x * 64;
  const int tid  = threadIdx.x;
  const int lane = tid & 31;
  const int wv   = tid >> 5;
  const int cw   = c0 + wv * 16;
  const int hlf  = lane >> 4;
  const int ln   = lane & 15;

  __shared__ __bf16 Vt[256 * 32];     // V^T chunk: Vt[h*32 + n]
  __shared__ __bf16 Pt[4][16 * 32];   // per-wave P tile, row-major [c][n]
#if HAVE_TDM
  __shared__ __bf16 Vraw[2][32 * 256];  // raw [n][h] chunks, TDM double buffer
#endif

  const __bf16* Qb = Q + ((size_t)b * 4096 + cw) * 256;
  const __bf16* Kb = K + (size_t)b * N * 256;
  const __bf16* Vb = V + (size_t)b * N * 256;
  const unsigned char* Mb = mask + (size_t)b * N * 4096;

  v16bf qf[8];
#pragma unroll
  for (int kh = 0; kh < 8; ++kh) qf[kh] = load_frag(Qb + kh * 32, 256, lane);

  v8f O[16];
#pragma unroll
  for (int t = 0; t < 16; ++t) O[t] = zero8();
  float m_run[8], l_run[8];
#pragma unroll
  for (int r = 0; r < 8; ++r) { m_run[r] = -3.0e38f; l_run[r] = 0.f; }

#if HAVE_TDM
  const unsigned vraw_off0 = lds_byte_off(&Vraw[0][0]);
  const unsigned vraw_off1 = lds_byte_off(&Vraw[1][0]);
  if (wv == 0) tdm_load_2d(vraw_off0, Vb, 32, 256, 256);  // chunk 0 in flight
#endif

  int bufsel = 0;
  for (int n0 = 0; n0 < N; n0 += 32, bufsel ^= 1) {
#if HAVE_TDM
    if (wv == 0) wait_tensorcnt0();     // chunk n0 landed in Vraw[bufsel]
    __syncthreads();
    if (wv == 0 && n0 + 32 < N)         // overlap next DMA with this compute
      tdm_load_2d(bufsel ? vraw_off0 : vraw_off1,
                  Vb + (size_t)(n0 + 32) * 256, 32, 256, 256);
    // LDS->LDS transpose: Vraw[n][h] -> Vt[h][n]
    const __bf16* Vsrc = &Vraw[bufsel][0];
#pragma unroll
    for (int it = 0; it < 8; ++it) {
      const int idx = tid + it * 128;
      const int n   = idx >> 5;
      const int hb  = (idx & 31) * 8;
      union { uint4 q; __bf16 e[8]; } u;
      u.q = *reinterpret_cast<const uint4*>(Vsrc + (size_t)n * 256 + hb);
#pragma unroll
      for (int j = 0; j < 8; ++j) Vt[(hb + j) * 32 + n] = u.e[j];
    }
#else
    // fallback: synchronous global->LDS transpose staging
#pragma unroll
    for (int it = 0; it < 8; ++it) {
      const int idx = tid + it * 128;
      const int n   = idx >> 5;
      const int hb  = (idx & 31) * 8;
      union { uint4 q; __bf16 e[8]; } u;
      u.q = *reinterpret_cast<const uint4*>(Vb + (size_t)(n0 + n) * 256 + hb);
#pragma unroll
      for (int j = 0; j < 8; ++j) Vt[(hb + j) * 32 + n] = u.e[j];
    }
#endif
    __syncthreads();

    if (n0 + 32 < N)                    // keep next K chunk warm (global_prefetch)
      __builtin_prefetch(Kb + (size_t)(n0 + 32 + lane) * 256, 0, 1);

    // --- scores S[c, n] for 32 keys, pipelined K-fragment loads ---
    v8f s0 = zero8(), s1 = zero8();
    v16bf k0c = load_frag(Kb + (size_t)n0 * 256, 256, lane);
    v16bf k1c = load_frag(Kb + (size_t)(n0 + 16) * 256, 256, lane);
#pragma unroll
    for (int kh = 0; kh < 8; ++kh) {
      v16bf k0n = k0c, k1n = k1c;
      if (kh < 7) {
        k0n = load_frag(Kb + (size_t)n0 * 256 + (kh + 1) * 32, 256, lane);
        k1n = load_frag(Kb + (size_t)(n0 + 16) * 256 + (kh + 1) * 32, 256, lane);
      }
      s0 = wmma_bf16(qf[kh], k0c, s0);
      s1 = wmma_bf16(qf[kh], k1c, s1);
      k0c = k0n; k1c = k1n;
    }

    // --- mask + online softmax over n (16-lane-group reduction) ---
    float p0[8], p1[8], scl[8];
#pragma unroll
    for (int r = 0; r < 8; ++r) {
      const int c  = cw + r + 8 * hlf;
      const int na = n0 + ln;
      const int nb = n0 + 16 + ln;
      float sa = s0[r], sb = s1[r];
      if (__builtin_nontemporal_load(Mb + (size_t)na * 4096 + c)) sa = -3.0e38f;
      if (__builtin_nontemporal_load(Mb + (size_t)nb * 4096 + c)) sb = -3.0e38f;
      float rm = fmaxf(sa, sb);
#pragma unroll
      for (int off = 1; off < 16; off <<= 1)
        rm = fmaxf(rm, __shfl_xor(rm, off, 32));
      const float mnew = fmaxf(m_run[r], rm);
      const float pa = __expf(sa - mnew);
      const float pb = __expf(sb - mnew);
      float rs = pa + pb;
#pragma unroll
      for (int off = 1; off < 16; off <<= 1) rs += __shfl_xor(rs, off, 32);
      const float sc = __expf(m_run[r] - mnew);
      l_run[r] = l_run[r] * sc + rs;
      m_run[r] = mnew;
      p0[r] = pa; p1[r] = pb; scl[r] = sc;
    }

#pragma unroll
    for (int t = 0; t < 16; ++t)
#pragma unroll
      for (int r = 0; r < 8; ++r) O[t][r] *= scl[r];

    // --- P (C-layout) -> LDS -> A fragment ---
#pragma unroll
    for (int r = 0; r < 8; ++r) {
      const int row = r + 8 * hlf;
      Pt[wv][row * 32 + ln]      = f2bf(p0[r]);
      Pt[wv][row * 32 + 16 + ln] = f2bf(p1[r]);
    }
    v16bf pf = load_frag(&Pt[wv][0], 32, lane);

    // --- O[c, h] += P(16x32) x V(32x16) per h-tile ---
#pragma unroll
    for (int t = 0; t < 16; ++t) {
      v16bf vf = load_frag(&Vt[(t * 16) * 32], 32, lane);
      O[t] = wmma_bf16(pf, vf, O[t]);
    }
    __syncthreads();
  }

  // epilogue: normalize, store bf16 context
#pragma unroll
  for (int r = 0; r < 8; ++r) {
    const float inv = (l_run[r] > 0.f) ? (1.0f / l_run[r]) : 0.f;
    const int c = cw + r + 8 * hlf;
    __bf16* outp = ctx + ((size_t)b * 4096 + c) * 256;
#pragma unroll
    for (int t = 0; t < 16; ++t) outp[t * 16 + ln] = f2bf(O[t][r] * inv);
  }
}

// ---------------------------------------------------------------------------
// K4: out[row,a] = tanh(sum_s Xs·Wl[a, s*256:...] + b[a]).  24 pipelined k-steps.
// ---------------------------------------------------------------------------
__global__ __launch_bounds__(128) void gemm_out_kernel(
    const __bf16* __restrict__ X0, const __bf16* __restrict__ X1,
    const __bf16* __restrict__ X2, const __bf16* __restrict__ Wl,
    const float* __restrict__ bias, float* __restrict__ out, int rows) {
  const int lane = threadIdx.x & 31;
  const int wv   = threadIdx.x >> 5;
  const int row0 = blockIdx.x * 64 + wv * 16;
  const int col0 = blockIdx.y * 64;
  if (row0 >= rows) return;

  v8f acc[4];
#pragma unroll
  for (int j = 0; j < 4; ++j) acc[j] = zero8();

  const __bf16* Xs[3] = {X0, X1, X2};
  v16bf a_cur = load_frag(X0 + (size_t)row0 * 256, 256, lane);
  v16bf b_cur[4];
#pragma unroll
  for (int j = 0; j < 4; ++j)
    b_cur[j] = load_frag(Wl + (size_t)(col0 + j * 16) * 768, 768, lane);

#pragma unroll
  for (int t = 0; t < 24; ++t) {
    v16bf a_nxt = a_cur;
    v16bf b_nxt[4];
    if (t < 23) {
      const int s2 = (t + 1) >> 3, kh2 = (t + 1) & 7;
      a_nxt = load_frag(Xs[s2] + (size_t)row0 * 256 + kh2 * 32, 256, lane);
#pragma unroll
      for (int j = 0; j < 4; ++j)
        b_nxt[j] = load_frag(
            Wl + (size_t)(col0 + j * 16) * 768 + s2 * 256 + kh2 * 32, 768, lane);
    }
#pragma unroll
    for (int j = 0; j < 4; ++j) acc[j] = wmma_bf16(a_cur, b_cur[j], acc[j]);
    a_cur = a_nxt;
#pragma unroll
    for (int j = 0; j < 4; ++j) b_cur[j] = (t < 23) ? b_nxt[j] : b_cur[j];
  }

  const int hlf = lane >> 4, ln = lane & 15;
#pragma unroll
  for (int j = 0; j < 4; ++j) {
    const int col = col0 + j * 16 + ln;
    const float bb = bias[col];
#pragma unroll
    for (int r = 0; r < 8; ++r) {
      const int row = row0 + r + 8 * hlf;
      __builtin_nontemporal_store(tanhf(acc[j][r] + bb),
                                  out + (size_t)row * 256 + col);
    }
  }
}

// ---------------------------------------------------------------------------
// Launch
// ---------------------------------------------------------------------------
extern "C" void kernel_launch(void* const* d_in, const int* in_sizes, int n_in,
                              void* d_out, int out_size, void* d_ws,
                              size_t ws_size, hipStream_t stream) {
  (void)in_sizes; (void)n_in; (void)out_size; (void)ws_size;

  const float* attendee_stmts = (const float*)d_in[0];   // [4,4096,256]
  const float* attendee_eres  = (const float*)d_in[1];   // [4,2048,256]
  const float* attender       = (const float*)d_in[2];   // [4,4096,256]
  const float* W_ss           = (const float*)d_in[3];   // [256,256]
  const float* b_ss           = (const float*)d_in[4];   // [256]
  const float* W_es           = (const float*)d_in[5];   // [256,256]
  const float* b_es           = (const float*)d_in[6];   // [256]
  const float* W_lin          = (const float*)d_in[7];   // [256,768]
  const float* b_lin          = (const float*)d_in[8];   // [256]
  const unsigned char* mask_ss = (const unsigned char*)d_in[9];   // [4,4096,4096]
  const unsigned char* mask_es = (const unsigned char*)d_in[10];  // [4,2048,4096]

  const int nQ  = 4 * 4096 * 256;
  const int nVs = 4 * 4096 * 256;
  const int nVe = 4 * 2048 * 256;
  const int nW  = 256 * 256;
  const int nWl = 256 * 768;

  char* ws = (char*)d_ws;
  size_t off = 0;
  auto alloc = [&](size_t bytes) {
    char* p = ws + off;
    off += (bytes + 255) & ~(size_t)255;
    return p;
  };
  __bf16* Qbf  = (__bf16*)alloc((size_t)nQ * 2);
  __bf16* Vs   = (__bf16*)alloc((size_t)nVs * 2);
  __bf16* Ve   = (__bf16*)alloc((size_t)nVe * 2);
  __bf16* Ks   = (__bf16*)alloc((size_t)nVs * 2);
  __bf16* Ke   = (__bf16*)alloc((size_t)nVe * 2);
  __bf16* Wss  = (__bf16*)alloc((size_t)nW * 2);
  __bf16* Wes  = (__bf16*)alloc((size_t)nW * 2);
  __bf16* Wlin = (__bf16*)alloc((size_t)nWl * 2);
  __bf16* Cs   = (__bf16*)alloc((size_t)nQ * 2);
  __bf16* Ce   = (__bf16*)alloc((size_t)nQ * 2);

  cvt_bf16_kernel<<<(nVs + 255) / 256, 256, 0, stream>>>(attendee_stmts, Vs, nVs);
  cvt_bf16_kernel<<<(nVe + 255) / 256, 256, 0, stream>>>(attendee_eres, Ve, nVe);
  cvt_bf16_kernel<<<(nQ + 255) / 256, 256, 0, stream>>>(attender, Qbf, nQ);
  cvt_bf16_kernel<<<(nW + 255) / 256, 256, 0, stream>>>(W_ss, Wss, nW);
  cvt_bf16_kernel<<<(nW + 255) / 256, 256, 0, stream>>>(W_es, Wes, nW);
  cvt_bf16_kernel<<<(nWl + 255) / 256, 256, 0, stream>>>(W_lin, Wlin, nWl);

  gemm_proj_kernel<<<dim3(16384 / 64, 4), 128, 0, stream>>>(Vs, Wss, b_ss, Ks, 16384);
  gemm_proj_kernel<<<dim3(8192 / 64, 4), 128, 0, stream>>>(Ve, Wes, b_es, Ke, 8192);

  flash_attend_kernel<<<dim3(4096 / 64, 4), 128, 0, stream>>>(Qbf, Ks, Vs, mask_ss,
                                                              Cs, 4096);
  flash_attend_kernel<<<dim3(4096 / 64, 4), 128, 0, stream>>>(Qbf, Ke, Ve, mask_es,
                                                              Ce, 2048);

  gemm_out_kernel<<<dim3(16384 / 64, 4), 128, 0, stream>>>(Qbf, Cs, Ce, Wlin, b_lin,
                                                           (float*)d_out, 16384);
}
